// LSTMModel_14791867368243
// MI455X (gfx1250) — matrix operand order repaired and checked by reading, not compiled, not run on hardware
//
#include <hip/hip_runtime.h>
#include <hip/hip_bf16.h>

// Problem constants
#define TT     200    // sequence length
#define BBATCH 256    // batch
#define EE     100    // embedding dim (real)
#define EPAD   128    // embedding dim padded to K multiple of 32
#define HH     512    // hidden
#define GG     2048   // 4*H (gates i,f,g,o)
#define NTILE  128    // GG / 16
#define NCLS   5
#define VPAD   50000  // pad token id
#define NBLOCKS 128u  // persistent grid: (HH/64) x (BBATCH/16) = 8 x 16

typedef __attribute__((ext_vector_type(16))) _Float16 v16h;
typedef __attribute__((ext_vector_type(8)))  _Float16 v8h;
typedef __attribute__((ext_vector_type(8)))  float    v8f;

union V16U { v16h v; v8h h[2]; };

__device__ __forceinline__ float fsigmoid(float x) {
    return __builtin_amdgcn_rcpf(1.0f + __builtin_amdgcn_exp2f(-1.44269504088896340736f * x));
}
__device__ __forceinline__ float ftanh(float x) {
    return 2.0f * __builtin_amdgcn_rcpf(1.0f + __builtin_amdgcn_exp2f(-2.88539008177779268147f * x)) - 1.0f;
}
__device__ __forceinline__ v8f wmma_f16(v16h a, v16h b, v8f c) {
    return __builtin_amdgcn_wmma_f32_16x16x32_f16(false, a, false, b, (short)0, c, false, false);
}

// Device-wide barrier (monotonic counter; grid of NBLOCKS fully resident).
// Release/acquire at agent scope => compiler emits the gfx1250 global_wb /
// global_inv sequences needed for cross-WGP h-state visibility.
__device__ __forceinline__ void grid_sync(unsigned* cnt, unsigned target) {
    __threadfence();          // each wave flushes its own h stores
    __syncthreads();
    if (threadIdx.x == 0) {
        __hip_atomic_fetch_add(cnt, 1u, __ATOMIC_RELEASE, __HIP_MEMORY_SCOPE_AGENT);
        while (__hip_atomic_load(cnt, __ATOMIC_RELAXED, __HIP_MEMORY_SCOPE_AGENT) < target) {
            __builtin_amdgcn_s_sleep(2);
        }
        (void)__hip_atomic_load(cnt, __ATOMIC_ACQUIRE, __HIP_MEMORY_SCOPE_AGENT);
    }
    __syncthreads();
}

// ---------------------------------------------------------------------------
// Embedding gather: xpad[t][b][k] = k < EE ? emb[X[b][t]][k] : 0   (f16)
// ---------------------------------------------------------------------------
__global__ void embed_kernel(const int* __restrict__ X,
                             const float* __restrict__ emb,
                             _Float16* __restrict__ xpad) {
    int idx = blockIdx.x * blockDim.x + threadIdx.x;   // over TT*BBATCH*EPAD
    if (idx >= TT * BBATCH * EPAD) return;
    int k = idx & (EPAD - 1);
    int b = (idx >> 7) & (BBATCH - 1);
    int t = idx >> 15;
    float v = 0.0f;
    if (k < EE) {
        int tok = X[b * TT + t];
        v = emb[tok * EE + k];
    }
    xpad[idx] = (_Float16)v;
}

// ---------------------------------------------------------------------------
// last[b] = max t with X[b][t] != pad, else -1 ; also zero barrier counters
// ---------------------------------------------------------------------------
__global__ void last_kernel(const int* __restrict__ X, int* __restrict__ last,
                            unsigned* __restrict__ bar0, unsigned* __restrict__ bar1) {
    int b = blockIdx.x * blockDim.x + threadIdx.x;
    if (b == 0) { *bar0 = 0u; *bar1 = 0u; }
    if (b >= BBATCH) return;
    int l = -1;
    for (int t = 0; t < TT; ++t)
        if (X[b * TT + t] != VPAD) l = t;
    last[b] = l;
}

// ---------------------------------------------------------------------------
// Pack Wcat = [W_ih.T ; W_hh.T] ([KTOT, GG]) into the CDNA5 WMMA B-matrix
// lane layout, f16. Flat: wpack[((kc*NTILE + nt)*32 + lane)*16 + e]
//   lane 0-15 : N = nt*16 + lane,     K = kc*32 + (e<8 ? e : 16+e-8)
//   lane 16-31: N = nt*16 + lane-16,  K = kc*32 + 8 + (e<8 ? e : 16+e-8)
// ---------------------------------------------------------------------------
__global__ void pack_kernel(const float* __restrict__ Wih,
                            const float* __restrict__ Whh,
                            _Float16* __restrict__ wpack,
                            int din, int KX, int KTOT) {
    int idx = blockIdx.x * blockDim.x + threadIdx.x;
    int total = (KTOT / 32) * NTILE * 32 * 16;
    if (idx >= total) return;
    int e    = idx & 15;
    int lane = (idx >> 4) & 31;
    int nt   = (idx >> 9) & (NTILE - 1);
    int kc   = idx >> 16;
    int gcol = nt * 16 + (lane & 15);
    int k    = kc * 32 + ((lane >= 16) ? 8 : 0) + ((e & 8) ? 16 : 0) + (e & 7);
    float v;
    if (k < KX) v = (k < din) ? Wih[gcol * din + k] : 0.0f;
    else        v = Whh[gcol * HH + (k - KX)];
    wpack[idx] = (_Float16)v;
}

// ---------------------------------------------------------------------------
// Persistent LSTM layer: whole T-loop in one kernel, grid barrier per step.
// Wave tile: 16 batch rows x 16 hidden cols; c (and pooled, layer 1) live in
// VGPRs for the entire sequence.
// Per step: x-phase GEMM first (no cross-block dep, hides barrier latency),
// then barrier, then h-phase GEMM. Both phases software-pipelined.
// ---------------------------------------------------------------------------
template <int KX, int KTOT, bool POOL>
__global__ __launch_bounds__(128) void lstm_persistent(
    const _Float16* __restrict__ xseq,   // [TT, BBATCH, KX] layer input
    _Float16*       __restrict__ hseq,   // layer0: [TT, BBATCH, HH] h output
    _Float16*       __restrict__ hA,     // layer1 ping-pong h buffers
    _Float16*       __restrict__ hB,
    const _Float16* __restrict__ wpack,
    const float*    __restrict__ bias,   // [GG]
    float*          __restrict__ pooled, // [BBATCH, HH] (layer1)
    const int*      __restrict__ lastpos,
    unsigned*       __restrict__ bar)
{
    constexpr int KCX = KX / 32;           // x-phase chunks
    constexpr int KCH = (KTOT - KX) / 32;  // h-phase chunks
    const int lane = threadIdx.x & 31;
    const int wave = threadIdx.x >> 5;
    const int hi16 = (lane >= 16) ? 1 : 0;
    const int j0   = blockIdx.x * 64 + wave * 16;   // hidden-col tile base
    const int b0   = blockIdx.y * 16;               // batch tile base
    const int jt   = j0 >> 4;
    const int m    = b0 + (lane & 15);              // A-operand batch row
    const int j    = j0 + (lane & 15);              // C/D hidden col (lane -> N)

    // kc-invariant, per-lane-half A sub-offsets (fold into load immediates)
    const int aoff0 = hi16 ? 8 : 0;
    const int aoff1 = hi16 ? 24 : 16;
    // per-lane base into packed weights for this wave's jt
    const _Float16* wlane = wpack + ((size_t)jt * 32 + lane) * 16;

    const float bias_i = bias[j];
    const float bias_f = bias[HH + j];
    const float bias_g = bias[2 * HH + j];
    const float bias_o = bias[3 * HH + j];

    v8f c_reg = {};          // cell state tile lives in registers across all t
    v8f pool_reg;
    int last8[8];
#pragma unroll
    for (int r = 0; r < 8; ++r) {
        pool_reg[r] = -3.402823466e38f;
        last8[r] = POOL ? lastpos[b0 + r + hi16 * 8] : 0;
    }

    for (int t = 0; t < TT; ++t) {
        v8f acc_i = {}, acc_f = {}, acc_g = {}, acc_o = {};

        // ---- software-pipelined GEMM over one contiguous operand region ----
        auto gemm_phase = [&](const _Float16* rowbase, int kcBase, int nchunks) {
            auto lA = [&](int kc) -> v16h {
                const _Float16* p = rowbase + kc * 32;
                V16U u;
                u.h[0] = *(const v8h*)(p + aoff0);
                u.h[1] = *(const v8h*)(p + aoff1);
                return u.v;
            };
            auto lB = [&](int kc, int g) -> v16h {
                return *(const v16h*)(wlane +
                    ((size_t)(kcBase + kc) * NTILE + g * 32) * (32 * 16));
            };
            v16h aC  = lA(0);
            v16h b0C = lB(0, 0), b1C = lB(0, 1), b2C = lB(0, 2), b3C = lB(0, 3);
#pragma unroll 2
            for (int kc = 0; kc + 1 < nchunks; ++kc) {
                v16h aN  = lA(kc + 1);
                v16h b0N = lB(kc + 1, 0);
                v16h b1N = lB(kc + 1, 1);
                v16h b2N = lB(kc + 1, 2);
                v16h b3N = lB(kc + 1, 3);
                acc_i = wmma_f16(aC, b0C, acc_i);
                acc_f = wmma_f16(aC, b1C, acc_f);
                acc_g = wmma_f16(aC, b2C, acc_g);
                acc_o = wmma_f16(aC, b3C, acc_o);
                aC = aN; b0C = b0N; b1C = b1N; b2C = b2N; b3C = b3N;
            }
            acc_i = wmma_f16(aC, b0C, acc_i);
            acc_f = wmma_f16(aC, b1C, acc_f);
            acc_g = wmma_f16(aC, b2C, acc_g);
            acc_o = wmma_f16(aC, b3C, acc_o);
        };

        // x-phase: depends only on read-only xseq -> run BEFORE the barrier
        gemm_phase(xseq + ((size_t)t * BBATCH + m) * KX, 0, KCX);

        // h-phase: needs all blocks' step t-1 h stores
        if (t > 0) {
            grid_sync(bar, (unsigned)t * NBLOCKS);
            const _Float16* hprev;
            if constexpr (POOL) hprev = ((t & 1) ? hA : hB) + (size_t)m * HH;
            else                hprev = hseq + ((size_t)(t - 1) * BBATCH + m) * HH;
            gemm_phase(hprev, KCX, KCH);
        }

        _Float16* hout;
        if constexpr (POOL) hout = (t & 1) ? hB : hA;
        else                hout = hseq + (size_t)t * BBATCH * HH;

#pragma unroll
        for (int r = 0; r < 8; ++r) {        // C/D: vgpr r -> M (batch row)
            const int brow = b0 + r + hi16 * 8;
            const float gi = fsigmoid(acc_i[r] + bias_i);
            const float gf = fsigmoid(acc_f[r] + bias_f);
            const float gg = ftanh   (acc_g[r] + bias_g);
            const float go = fsigmoid(acc_o[r] + bias_o);
            const float cnew = gf * c_reg[r] + gi * gg;
            const float h    = go * ftanh(cnew);
            c_reg[r] = cnew;
            hout[(size_t)brow * HH + j] = (_Float16)h;
            if constexpr (POOL) {
                if (t <= last8[r]) pool_reg[r] = fmaxf(pool_reg[r], h);
            }
        }
    }

    if constexpr (POOL) {
#pragma unroll
        for (int r = 0; r < 8; ++r)
            pooled[(size_t)(b0 + r + hi16 * 8) * HH + j] = pool_reg[r];
    }
}

// ---------------------------------------------------------------------------
// logits[b][cls] = pooled[b] . W_out[cls] + b_out[cls]
// ---------------------------------------------------------------------------
__global__ void logits_kernel(const float* __restrict__ pooled,
                              const float* __restrict__ Wout,
                              const float* __restrict__ bout,
                              float* __restrict__ out) {
    int idx = blockIdx.x * blockDim.x + threadIdx.x;
    if (idx >= BBATCH * NCLS) return;
    int cls = idx % NCLS;
    int b   = idx / NCLS;
    float s = bout[cls];
    for (int jj = 0; jj < HH; ++jj)
        s += pooled[b * HH + jj] * Wout[cls * HH + jj];
    out[b * NCLS + cls] = s;
}

// ---------------------------------------------------------------------------
extern "C" void kernel_launch(void* const* d_in, const int* in_sizes, int n_in,
                              void* d_out, int out_size, void* d_ws, size_t ws_size,
                              hipStream_t stream) {
    (void)in_sizes; (void)n_in; (void)out_size; (void)ws_size;

    const int*   X    = (const int*)  d_in[0];
    const float* emb  = (const float*)d_in[1];
    const float* Wih0 = (const float*)d_in[2];
    const float* Whh0 = (const float*)d_in[3];
    const float* b0v  = (const float*)d_in[4];
    const float* Wih1 = (const float*)d_in[5];
    const float* Whh1 = (const float*)d_in[6];
    const float* b1v  = (const float*)d_in[7];
    const float* Wout = (const float*)d_in[8];
    const float* bout = (const float*)d_in[9];
    float* out = (float*)d_out;

    char* ws = (char*)d_ws;
    size_t off = 0;
    auto alloc = [&](size_t bytes) -> void* {
        void* p = ws + off;
        off = (off + bytes + 255) & ~(size_t)255;
        return p;
    };

    _Float16* xpad   = (_Float16*)alloc((size_t)TT * BBATCH * EPAD * 2);  // 13.1 MB
    _Float16* hseq0  = (_Float16*)alloc((size_t)TT * BBATCH * HH * 2);    // 52.4 MB
    _Float16* hbufA  = (_Float16*)alloc((size_t)BBATCH * HH * 2);
    _Float16* hbufB  = (_Float16*)alloc((size_t)BBATCH * HH * 2);
    float*    pooled = (float*)   alloc((size_t)BBATCH * HH * 4);
    int*      last   = (int*)     alloc((size_t)BBATCH * 4);
    unsigned* bar0   = (unsigned*)alloc(256);
    unsigned* bar1   = (unsigned*)alloc(256);
    const int KTOT0 = EPAD + HH;   // 640
    const int KTOT1 = HH + HH;     // 1024
    _Float16* wpack0 = (_Float16*)alloc((size_t)(KTOT0 / 32) * NTILE * 32 * 16 * 2);
    _Float16* wpack1 = (_Float16*)alloc((size_t)(KTOT1 / 32) * NTILE * 32 * 16 * 2);

    // --- preprocessing ---
    {
        int n = TT * BBATCH * EPAD;
        embed_kernel<<<(n + 255) / 256, 256, 0, stream>>>(X, emb, xpad);
    }
    last_kernel<<<1, 256, 0, stream>>>(X, last, bar0, bar1);
    {
        int n0 = (KTOT0 / 32) * NTILE * 32 * 16;
        pack_kernel<<<(n0 + 255) / 256, 256, 0, stream>>>(Wih0, Whh0, wpack0, EE, EPAD, KTOT0);
        int n1 = (KTOT1 / 32) * NTILE * 32 * 16;
        pack_kernel<<<(n1 + 255) / 256, 256, 0, stream>>>(Wih1, Whh1, wpack1, HH, HH, KTOT1);
    }

    dim3 grid(HH / 64, BBATCH / 16);   // (8, 16) = NBLOCKS, fully resident
    dim3 block(128);

    // --- layer 0: persistent recurrence, h_t -> hseq0[t] ---
    lstm_persistent<EPAD, EPAD + HH, false><<<grid, block, 0, stream>>>(
        xpad, hseq0, hbufA, hbufB, wpack0, b0v, pooled, last, bar0);

    // --- layer 1: persistent recurrence + fused masked max-pool ---
    lstm_persistent<HH, HH + HH, true><<<grid, block, 0, stream>>>(
        hseq0, hseq0, hbufA, hbufB, wpack1, b1v, pooled, last, bar1);

    // --- head ---
    {
        int n = BBATCH * NCLS;
        logits_kernel<<<(n + 255) / 256, 256, 0, stream>>>(pooled, Wout, bout, out);
    }
}
